// LayerConvolSelf_85418309583435
// MI455X (gfx1250) — matrix operand order repaired and checked by reading, not compiled, not run on hardware
//
#include <hip/hip_runtime.h>
#include <hip/hip_bf16.h>

// Problem constants (match reference)
#define B_   16
#define M_   100000
#define D_   128
#define OUT_ 128

typedef __attribute__((ext_vector_type(16))) __bf16        v16bf;
typedef __attribute__((ext_vector_type(8)))  float         v8f;
typedef __attribute__((ext_vector_type(8)))  unsigned int  v8u;

// ---------- helpers ----------
// pack two f32 -> two bf16 (RNE) in one uint; use native v_cvt_pk_bf16_f32
#if defined(__HIP_DEVICE_COMPILE__)
#if __has_builtin(__builtin_amdgcn_cvt_pk_bf16_f32)
__device__ __forceinline__ unsigned pk(float lo, float hi) {
  auto p = __builtin_amdgcn_cvt_pk_bf16_f32(lo, hi);   // 2 x bf16
  return __builtin_bit_cast(unsigned, p);
}
#else
__device__ __forceinline__ unsigned pk(float lo, float hi) {
  unsigned r;
  asm("v_cvt_pk_bf16_f32 %0, %1, %2" : "=v"(r) : "v"(lo), "v"(hi));
  return r;
}
#endif
#else
// host-pass stub (never executed on device)
__device__ __forceinline__ unsigned pk(float lo, float hi) {
  unsigned ul = __float_as_uint(lo), uh = __float_as_uint(hi);
  ul = (ul + 0x7FFFu + ((ul >> 16) & 1u)) >> 16;
  uh = (uh + 0x7FFFu + ((uh >> 16) & 1u)) >> 16;
  return ul | (uh << 16);
}
#endif
// order-preserving float<->uint map (for atomicMax on floats)
__device__ __forceinline__ unsigned f2ord(float f) {
  unsigned u = __float_as_uint(f);
  return (u & 0x80000000u) ? ~u : (u | 0x80000000u);
}
__device__ __forceinline__ float ord2f(unsigned u) {
  return (u & 0x80000000u) ? __uint_as_float(u & 0x7FFFFFFFu) : __uint_as_float(~u);
}
__device__ __forceinline__ v16bf load8u(const unsigned* p) {
  v8u u;
  #pragma unroll
  for (int q = 0; q < 8; ++q) u[q] = p[q];
  return __builtin_bit_cast(v16bf, u);
}

// ---------- K1: anchor_part + init accumulators ----------
__global__ void __launch_bounds__(256) k1_init_anchor(
    const float* __restrict__ xq, const float* __restrict__ W1,
    float* __restrict__ anchor, unsigned* __restrict__ maxb,
    float* __restrict__ sumexp, float* __restrict__ vacc)
{
  int tid = threadIdx.x;
  if (tid < B_) { maxb[tid] = 0u; sumexp[tid] = 0.f; }
  #pragma unroll
  for (int j = 0; j < 8; ++j) vacc[tid + 256 * j] = 0.f;
  // anchor_part[b][o] = xq[b,:] . W1[o, 0:128]
  #pragma unroll 1
  for (int j = 0; j < 8; ++j) {
    int idx = tid + 256 * j;            // 0..2047
    int b = idx >> 7, o = idx & 127;
    const float* xr = xq + b * D_;
    const float* wr = W1 + o * (2 * D_);
    float acc = 0.f;
    for (int k = 0; k < D_; ++k) acc = fmaf(xr[k], wr[k], acc);
    anchor[idx] = acc;
  }
}

// ---------- K2: node GEMM (WMMA bf16) + relu.W2 epilogue -> scores, running max ----------
__global__ void __launch_bounds__(256) k2_scores(
    const float* __restrict__ input, const float* __restrict__ W1,
    const float* __restrict__ W2, const float* __restrict__ anchor,
    float* __restrict__ scores, unsigned* __restrict__ maxb)
{
  // W1b (bf16) in B-layout: [ot(8)][s(4)][lane(32)][8 uints] = 8192 uints = 32KB
  __shared__ __align__(16) unsigned ldsB[8192];
  __shared__ float ldsA[2048];          // anchor_part [b][o], 8KB
  int tid  = threadIdx.x;
  int wave = tid >> 5, lane = tid & 31;
  int ln = lane & 15, kg = lane >> 4;
  int t = blockIdx.x * 8 + wave;        // 16-row tile index

  // prefetch this wave's A-tile rows so HBM fetch overlaps LDS staging
  if (t < (M_ / 16)) {
    const float* pr = input + (size_t)(t * 16 + ln) * D_;   // one 512B row per lane
    __builtin_prefetch(pr + kg * 64, 0, 1);                 // bytes 0 / 256
    __builtin_prefetch(pr + 32 + kg * 64, 0, 1);            // bytes 128 / 384
  }

  #pragma unroll
  for (int j = 0; j < 8; ++j) { int i = tid + 256 * j; ldsA[i] = anchor[i]; }

  // stage W1b: B element (k,o) = W1[o, 128+k]; lanes 0-15 hold K=kg*16..+15 contiguous
  #pragma unroll 1
  for (int j = 0; j < 4; ++j) {
    int trip  = tid + 256 * j;          // 0..1023
    int lslot = trip & 31;
    int s     = (trip >> 5) & 3;
    int ot    = trip >> 7;
    int o     = ot * 16 + (lslot & 15);
    int kgs   = lslot >> 4;
    const float* wr = W1 + o * (2 * D_) + D_ + s * 32 + kgs * 16;  // 16 contiguous floats
    unsigned* dst = ldsB + ((ot * 4 + s) * 32 + lslot) * 8;
    #pragma unroll
    for (int q = 0; q < 4; ++q) {
      float4 f = *(const float4*)(wr + 4 * q);
      dst[2 * q]     = pk(f.x, f.y);
      dst[2 * q + 1] = pk(f.z, f.w);
    }
  }
  __syncthreads();

  if (t >= (M_ / 16)) return;

  float w2v[8];
  #pragma unroll
  for (int ot = 0; ot < 8; ++ot) w2v[ot] = W2[ot * 16 + ln];

  // A tiles: input rows, A operand, K order as in ldsB (lane halves hold K kg*16..+15 contiguous)
  v16bf a[4];
  const float* arow = input + (size_t)(t * 16 + ln) * D_;
  #pragma unroll
  for (int s = 0; s < 4; ++s) {
    const float* p0 = arow + s * 32 + kg * 16;
    const float* p1 = p0 + 8;
    float4 q0 = *(const float4*)(p0);
    float4 q1 = *(const float4*)(p0 + 4);
    float4 q2 = *(const float4*)(p1);
    float4 q3 = *(const float4*)(p1 + 4);
    v8u u;
    u[0] = pk(q0.x, q0.y); u[1] = pk(q0.z, q0.w);
    u[2] = pk(q1.x, q1.y); u[3] = pk(q1.z, q1.w);
    u[4] = pk(q2.x, q2.y); u[5] = pk(q2.z, q2.w);
    u[6] = pk(q3.x, q3.y); u[7] = pk(q3.z, q3.w);
    a[s] = __builtin_bit_cast(v16bf, u);
  }

  // node_part tiles: C[ot] = input_tile @ W1b.T (o-tile ot), f32 accum
  v8f c[8];
  #pragma unroll
  for (int ot = 0; ot < 8; ++ot) {
    v8f acc = {0.f, 0.f, 0.f, 0.f, 0.f, 0.f, 0.f, 0.f};
    #pragma unroll
    for (int s = 0; s < 4; ++s) {
      v16bf bm = load8u(ldsB + ((ot * 4 + s) * 32 + lane) * 8);
      acc = __builtin_amdgcn_wmma_f32_16x16x32_bf16(false, a[s], false, bm,
                                                    (short)0, acc, false, false);
    }
    c[ot] = acc;
  }

  // epilogue: score[b][m] = sum_o relu(anchor[b,o] + node[m,o]) * W2[o]
  int m0 = t * 16;
  #pragma unroll 1
  for (int b = 0; b < B_; ++b) {
    float s8[8] = {0, 0, 0, 0, 0, 0, 0, 0};
    #pragma unroll
    for (int ot = 0; ot < 8; ++ot) {
      float anc = ldsA[b * 128 + ot * 16 + ln];
      #pragma unroll
      for (int r = 0; r < 8; ++r) {
        float h = fmaxf(anc + c[ot][r], 0.f);
        s8[r] = fmaf(h, w2v[ot], s8[r]);
      }
    }
    #pragma unroll
    for (int r = 0; r < 8; ++r) {
      #pragma unroll
      for (int msk = 1; msk < 16; msk <<= 1)
        s8[r] += __shfl_xor(s8[r], msk, 32);   // reduce over 16 o-lanes per half
    }
    float mx = s8[0];
    #pragma unroll
    for (int r = 1; r < 8; ++r) mx = fmaxf(mx, s8[r]);
    mx = fmaxf(mx, __shfl_xor(mx, 16, 32));
    if (lane == 0) atomicMax(maxb + b, f2ord(mx));
    if (ln == 0) {
      float* dst = scores + (size_t)b * M_ + m0 + kg * 8;   // half kg holds m = kg*8+r
      #pragma unroll
      for (int r = 0; r < 8; ++r) dst[r] = s8[r];
    }
  }
}

// ---------- K3: e = exp(s - max); vacc += e @ input (WMMA bf16); sumexp ----------
__global__ void __launch_bounds__(128) k3_pool(
    const float* __restrict__ input, const float* __restrict__ scores,
    const unsigned* __restrict__ maxb, float* __restrict__ sumexp,
    float* __restrict__ vacc)
{
  // per-wave B stage: [dt(8)][lane(32)][16 halves] = 4096 ushorts = 8KB each (4 waves)
  __shared__ __align__(16) unsigned short ldsBS[4][4096];
  __shared__ float ldsV[2048];          // block-level vacc accumulator, 8KB
  int tid = threadIdx.x;
  int wave = tid >> 5, lane = tid & 31;
  int ln = lane & 15, kg = lane >> 4;
  int t = blockIdx.x * 4 + wave;        // 32-row tile index
  const int T3 = M_ / 32;               // 3125

  // prefetch this wave's input tile + score chunks (overlap with LDS zeroing)
  if (t < T3) {
    const float* pr = input + (size_t)(t * 32 + lane) * D_;  // one 512B row per lane
    __builtin_prefetch(pr, 0, 1);
    __builtin_prefetch(pr + 32, 0, 1);
    __builtin_prefetch(pr + 64, 0, 1);
    __builtin_prefetch(pr + 96, 0, 1);
    __builtin_prefetch(scores + (size_t)ln * M_ + t * 32 + kg * 16, 0, 1);
  }

  #pragma unroll
  for (int j = 0; j < 16; ++j) ldsV[tid + 128 * j] = 0.f;
  __syncthreads();

  if (t < T3) {
    int m0 = t * 32;
    // stage input tile [32m x 128d] -> bf16 in B layout (lane=d, elem=K within 16)
    unsigned short* wb = &ldsBS[wave][0];
    int dt    = lane >> 2;              // this lane's 4 columns all live in d-tile dt
    int lbase = (lane & 3) * 4;         // d&15 for jj=0
    #pragma unroll 4
    for (int i = 0; i < 32; ++i) {
      float4 f = *(const float4*)(input + (size_t)(m0 + i) * D_ + lane * 4);
      int half = (i >> 4) << 4;         // K>=16 goes to lane-slots 16..31
      int elem = i & 15;
      unsigned p01 = pk(f.x, f.y);
      unsigned p23 = pk(f.z, f.w);
      unsigned short* base = wb + dt * 512 + (lbase + half) * 16 + elem;
      base[0 * 16] = (unsigned short)(p01 & 0xFFFFu);
      base[1 * 16] = (unsigned short)(p01 >> 16);
      base[2 * 16] = (unsigned short)(p23 & 0xFFFFu);
      base[3 * 16] = (unsigned short)(p23 >> 16);
    }

    // A = exp weights [16b x 32m]; lane&15 = b, halves hold m kg*16..+15 contiguous
    const float* srow = scores + (size_t)ln * M_ + m0;
    float mx = ord2f(maxb[ln]);
    float4 e0 = *(const float4*)(srow + kg * 16);
    float4 e1 = *(const float4*)(srow + kg * 16 + 4);
    float4 e2 = *(const float4*)(srow + kg * 16 + 8);
    float4 e3 = *(const float4*)(srow + kg * 16 + 12);
    float e[16] = {e0.x, e0.y, e0.z, e0.w, e1.x, e1.y, e1.z, e1.w,
                   e2.x, e2.y, e2.z, e2.w, e3.x, e3.y, e3.z, e3.w};
    float mysum = 0.f;
    #pragma unroll
    for (int q = 0; q < 16; ++q) { e[q] = __expf(e[q] - mx); mysum += e[q]; }
    mysum += __shfl_xor(mysum, 16, 32);
    if (lane < 16) atomicAdd(&sumexp[ln], mysum);

    v8u ua;
    #pragma unroll
    for (int q = 0; q < 8; ++q) ua[q] = pk(e[2 * q], e[2 * q + 1]);
    v16bf ea = __builtin_bit_cast(v16bf, ua);

    // vacc_tile[16b x 16d] += E[16x32] @ input[32x16] per d-tile
    #pragma unroll
    for (int dd = 0; dd < 8; ++dd) {
      v16bf bm = load8u((const unsigned*)(wb + dd * 512 + lane * 16));
      v8f acc = {0.f, 0.f, 0.f, 0.f, 0.f, 0.f, 0.f, 0.f};
      acc = __builtin_amdgcn_wmma_f32_16x16x32_bf16(false, ea, false, bm,
                                                    (short)0, acc, false, false);
      #pragma unroll
      for (int r = 0; r < 8; ++r)       // lane holds b=r+8*kg, d=dd*16+ln
        atomicAdd(&ldsV[(r + 8 * kg) * 128 + dd * 16 + ln], acc[r]);
    }
  }
  __syncthreads();
  #pragma unroll
  for (int j = 0; j < 16; ++j) {
    int i = tid + 128 * j;
    atomicAdd(&vacc[i], ldsV[i]);
  }
}

// ---------- K4: normalize ----------
__global__ void __launch_bounds__(256) k4_final(
    const float* __restrict__ vacc, const float* __restrict__ sumexp,
    float* __restrict__ out)
{
  int tid = threadIdx.x;
  #pragma unroll
  for (int j = 0; j < 8; ++j) {
    int i = tid + 256 * j;              // 0..2047
    out[i] = vacc[i] / sumexp[i >> 7];
  }
}

extern "C" void kernel_launch(void* const* d_in, const int* in_sizes, int n_in,
                              void* d_out, int out_size, void* d_ws, size_t ws_size,
                              hipStream_t stream) {
  const float* xq    = (const float*)d_in[0];   // [16,128]
  const float* input = (const float*)d_in[1];   // [100000,128]
  // d_in[2] = adj (unused in forward math)
  const float* W1    = (const float*)d_in[3];   // [128,256]
  const float* W2    = (const float*)d_in[4];   // [1,128]

  float*    ws     = (float*)d_ws;
  float*    anchor = ws;                         // 2048 f32
  unsigned* maxb   = (unsigned*)(ws + 2048);     // 16 ordered-uint
  float*    sumexp = ws + 2064;                  // 16 f32
  float*    vacc   = ws + 2080;                  // 2048 f32
  float*    scores = ws + 4128;                  // 16*100000 f32 (~6.4MB)

  k1_init_anchor<<<dim3(1), dim3(256), 0, stream>>>(xq, W1, anchor, maxb, sumexp, vacc);

  int blocks2 = (M_ / 16 + 7) / 8;               // 782 blocks x 8 waves x 16 rows
  k2_scores<<<dim3(blocks2), dim3(256), 0, stream>>>(input, W1, W2, anchor, scores, maxb);

  int blocks3 = (M_ / 32 + 3) / 4;               // 782 blocks x 4 waves x 32 rows
  k3_pool<<<dim3(blocks3), dim3(128), 0, stream>>>(input, scores, maxb, sumexp, vacc);

  k4_final<<<dim3(1), dim3(256), 0, stream>>>(vacc, sumexp, (float*)d_out);
}
